// MultiHeadSelfAttention_12137577578770
// MI455X (gfx1250) — matrix operand order repaired and checked by reading, not compiled, not run on hardware
//
#include <hip/hip_runtime.h>

// ---------------------------------------------------------------------------
// LoRA multi-head self-attention forward for MI455X (gfx1250), bf16 WMMA path.
//   W_eff = W + 0.5*A*B (rank-16) -> plain GEMMs + flash-style attention.
//   CDNA5 paths: v_wmma_f32_16x16x32_bf16, global_load_async_to_lds_b128
//   (ASYNCcnt, double-buffered), tensor_load_to_lds (TDM), global_prefetch.
// ---------------------------------------------------------------------------

typedef __bf16 bf16_t;
typedef __attribute__((ext_vector_type(16))) __bf16 v16bf;
typedef __attribute__((ext_vector_type(8)))  __bf16 v8bf;
typedef __attribute__((ext_vector_type(8)))  float  v8f;
typedef __attribute__((ext_vector_type(4)))  unsigned int u32x4;
typedef __attribute__((ext_vector_type(8)))  int i32x8;
typedef __attribute__((ext_vector_type(4)))  int i32x4;

#define DIM     2048
#define HEADS   16
#define HD      128
#define RANK    16
#define SEQ     2048
#define BATCH   4
#define MTOT    (BATCH * SEQ)          // 8192 rows
#define ATT_SCALE 0.08838834764831845f // 128^-0.5

__device__ __forceinline__ v16bf cat16(v8bf lo, v8bf hi) {
  v16bf r;
#pragma unroll
  for (int e = 0; e < 8; ++e) { r[e] = lo[e]; r[e + 8] = hi[e]; }
  return r;
}

// async global -> LDS, 16 bytes per lane (GLOBAL_LOAD_ASYNC_TO_LDS_B128)
__device__ __forceinline__ void async_copy16(const bf16_t* g, bf16_t* l) {
  unsigned int lds = (unsigned int)(uintptr_t)l;  // low 32 bits = LDS offset
  asm volatile("global_load_async_to_lds_b128 %0, %1, off"
               :: "v"(lds), "v"(g) : "memory");
}
__device__ __forceinline__ void wait_async0() {
  asm volatile("s_wait_asynccnt 0x0" ::: "memory");
}

// ---------------------------------------------------------------------------
// Kernel 1: x (f32) -> bf16
// ---------------------------------------------------------------------------
__global__ __launch_bounds__(256) void convert_f32_bf16(
    const float* __restrict__ x, bf16_t* __restrict__ y, int n) {
  int i = blockIdx.x * 256 + threadIdx.x;
  if (i < n) y[i] = (bf16_t)x[i];
}

// ---------------------------------------------------------------------------
// Kernel 2: swizzled effective weight:
//   WT[((d>>5)*DIM + o)*32 + (d&31)] = bf16( W[o][d] + 0.5*sum_r A[o][r]B[r][d] )
// -> GEMM B-tiles (128 n-rows x 32 k) are 64B-contiguous per row: pure copies.
// ---------------------------------------------------------------------------
__global__ __launch_bounds__(256) void prep_weight(
    const float* __restrict__ W, const float* __restrict__ A,
    const float* __restrict__ Bm, bf16_t* __restrict__ WT) {
  int idx = blockIdx.x * 256 + threadIdx.x;   // o*DIM + d (coalesced reads)
  int o = idx >> 11;
  int d = idx & (DIM - 1);
  float acc = W[(size_t)o * DIM + d];
#pragma unroll
  for (int r = 0; r < RANK; ++r)
    acc += 0.5f * A[o * RANK + r] * Bm[r * DIM + d];
  WT[((size_t)(d >> 5) * DIM + o) * 32 + (d & 31)] = (bf16_t)acc;
}

// ---------------------------------------------------------------------------
// Kernel 3: GEMM  Y[m][o] = sum_d X[m][d]*Weff[o][d] + bias[o]   (bf16 WMMA)
//  MODE 0: out bf16, head-split [b][h][l][hd], scaled by `scale`
//  MODE 1: out f32, plain [m][o]
//  MODE 2: out bf16, head-split transposed [b][h][hd][l]  (for V)
//  Workgroup 128x128, 8 waves (4M x 2N), wave tile 32x64, K-step 32.
//  Double-buffered LDS staged with global_load_async_to_lds_b128 (ASYNCcnt).
// ---------------------------------------------------------------------------
template <int MODE>
__global__ __launch_bounds__(256) void gemm_bias(
    const bf16_t* __restrict__ X,    // [MTOT][DIM]
    const bf16_t* __restrict__ WTs,  // swizzled [DIM/32][DIM][32]
    const float*  __restrict__ bias,
    bf16_t* __restrict__ out_bf, float* __restrict__ out_f, float scale) {
  __shared__ __align__(16) bf16_t As[2][128][40];  // [m][k], 80B row stride
  __shared__ __align__(16) bf16_t Bs[2][128][40];  // [n][k]

  const int n0 = blockIdx.x * 128;
  const int m0 = blockIdx.y * 128;
  const int tid  = threadIdx.x;
  const int wave = tid >> 5;
  const int lane = tid & 31;
  const int lrow = lane & 15;
  const int lhalf = lane >> 4;
  const int wm = wave >> 1;  // 0..3 -> 32 rows
  const int wn = wave & 1;   // 0..1 -> 64 cols
  const int r2 = tid >> 1;          // copy row 0..127
  const int c2 = (tid & 1) * 16;    // copy col 0 / 16

  v8f acc[2][4];
#pragma unroll
  for (int i = 0; i < 2; ++i)
#pragma unroll
    for (int j = 0; j < 4; ++j) acc[i][j] = (v8f){0.f,0.f,0.f,0.f,0.f,0.f,0.f,0.f};

  auto issue_tile = [&](int k0, int buf) {
    const bf16_t* asrc = X + (size_t)(m0 + r2) * DIM + k0 + c2;
    async_copy16(asrc,     &As[buf][r2][c2]);
    async_copy16(asrc + 8, &As[buf][r2][c2 + 8]);
    const bf16_t* bsrc = WTs + ((size_t)(k0 >> 5) * DIM + n0 + r2) * 32 + c2;
    async_copy16(bsrc,     &Bs[buf][r2][c2]);
    async_copy16(bsrc + 8, &Bs[buf][r2][c2 + 8]);
  };

  issue_tile(0, 0);
  wait_async0();
  __syncthreads();

  const int NT = DIM / 32;  // 64 k-tiles
  for (int t = 0; t < NT; ++t) {
    const int buf = t & 1;
    if (t + 1 < NT) issue_tile((t + 1) * 32, buf ^ 1);  // overlap DMA w/ WMMA

    v16bf af[2];
#pragma unroll
    for (int i = 0; i < 2; ++i) {
      int row = wm * 32 + i * 16 + lrow;
      int kb  = lhalf * 8;  // A layout: K kb..kb+7 then kb+16..kb+23
      af[i] = cat16(*(const v8bf*)&As[buf][row][kb],
                    *(const v8bf*)&As[buf][row][kb + 16]);
    }
    v16bf bfrag[4];
#pragma unroll
    for (int j = 0; j < 4; ++j) {
      int col = wn * 64 + j * 16 + lrow;
      int kb  = lhalf * 8;   // B layout: K kb..kb+7 then kb+16..kb+23
      bfrag[j] = cat16(*(const v8bf*)&Bs[buf][col][kb],
                       *(const v8bf*)&Bs[buf][col][kb + 16]);
    }
#pragma unroll
    for (int i = 0; i < 2; ++i)
#pragma unroll
      for (int j = 0; j < 4; ++j)
        acc[i][j] = __builtin_amdgcn_wmma_f32_16x16x32_bf16(
            false, af[i], false, bfrag[j], (short)0, acc[i][j], false, false);

    if (t + 1 < NT) wait_async0();  // next tile landed (in-order completion)
    __syncthreads();                // all waves done reading buf
  }

  // epilogue: C layout -> lane holds col lrow, rows lhalf*8 + v
#pragma unroll
  for (int i = 0; i < 2; ++i) {
#pragma unroll
    for (int j = 0; j < 4; ++j) {
      int colg = n0 + wn * 64 + j * 16 + lrow;
      float bv = bias[colg];
#pragma unroll
      for (int v = 0; v < 8; ++v) {
        int rowg  = m0 + wm * 32 + i * 16 + lhalf * 8 + v;
        float val = (acc[i][j][v] + bv) * scale;
        int b = rowg >> 11, l = rowg & (SEQ - 1);
        int h = colg >> 7,  hd = colg & (HD - 1);
        if constexpr (MODE == 0) {
          out_bf[(((size_t)(b * HEADS + h)) * SEQ + l) * HD + hd] = (bf16_t)val;
        } else if constexpr (MODE == 2) {
          out_bf[(((size_t)(b * HEADS + h)) * HD + hd) * SEQ + l] = (bf16_t)val;
        } else {
          out_f[(size_t)rowg * DIM + colg] = val;
        }
      }
    }
  }
}

// ---------------------------------------------------------------------------
// Kernel 4: attention per (b,h), 128-query block, flash-style, Q pre-scaled.
//  Q block staged by TDM (tensor_load_to_lds) with LDS padding matching
//  Qs[128][136]. Pass 1 keeps per-lane online softmax partials (no cross-lane
//  ops in the loop); one associative shuffle-combine at the end.
//  V is pre-transposed [b][h][hd][l] so P*V B-fragments load from global.
// ---------------------------------------------------------------------------
__global__ __launch_bounds__(256) void attention(
    const bf16_t* __restrict__ Q,   // [B*H][SEQ][HD]
    const bf16_t* __restrict__ K,   // [B*H][SEQ][HD]
    const bf16_t* __restrict__ Vt,  // [B*H][HD][SEQ]  (transposed)
    bf16_t* __restrict__ O) {       // [MTOT][DIM] merged heads
  __shared__ __align__(16) bf16_t Qs[128][136];   // [qrow][hd] (+8 pad)
  __shared__ __align__(16) bf16_t Ps[8][16][40];  // per-wave P [row][key]

  const int bh = blockIdx.y;
  const int q0 = blockIdx.x * 128;
  const bf16_t* Qb  = Q  + (size_t)bh * SEQ * HD;
  const bf16_t* Kb  = K  + (size_t)bh * SEQ * HD;
  const bf16_t* Vtb = Vt + (size_t)bh * HD * SEQ;

  const int tid  = threadIdx.x;
  const int wave = tid >> 5;
  const int lane = tid & 31;
  const int lrow = lane & 15;
  const int lhalf = lane >> 4;
  const int qrow = wave * 16;

  // ---- stage Q block (128x128) via TDM ------------------------------------
  if (wave == 0) {
    unsigned long long ga = (unsigned long long)(uintptr_t)(Qb + (size_t)q0 * HD);
    u32x4 g0;
    g0[0] = 1u;                                    // count=1
    g0[1] = (unsigned int)(uintptr_t)&Qs[0][0];    // lds_addr
    g0[2] = (unsigned int)ga;                      // global_addr[31:0]
    g0[3] = (unsigned int)((ga >> 32) & 0x01FFFFFFu) | 0x80000000u;  // type=2
    i32x8 g1;
    g1[0] = 0x07510000;         // data_size=2B, pad_en, interval=64dw, pad=4dw
    g1[1] = (int)(128u << 16);  // tensor_dim0 = 128
    g1[2] = (int)(2048u << 16); // tensor_dim1 = 2048
    g1[3] = (int)(128u << 16);  // tile_dim0 = 128
    g1[4] = 128;                // tile_dim1 = 128, tile_dim2 = 0
    g1[5] = 128;                // tensor_dim0_stride = 128
    g1[6] = 0;
    g1[7] = 0;
    i32x4 gz4 = {0, 0, 0, 0};
    i32x8 gz8 = {0, 0, 0, 0, 0, 0, 0, 0};
    __builtin_amdgcn_tensor_load_to_lds(g0, g1, gz4, gz4, gz8, 0);
    __builtin_amdgcn_s_wait_tensorcnt(0);
  }
  __syncthreads();

  // ---- pass 1: per-lane online max/sumexp over this lane's column stream --
  float mrow[8], srow[8];
#pragma unroll
  for (int v = 0; v < 8; ++v) { mrow[v] = -3.0e38f; srow[v] = 0.f; }

  for (int n0 = 0; n0 < SEQ; n0 += 16) {
    __builtin_prefetch(Kb + (size_t)(n0 + 16 + lrow) * HD, 0, 0);
    v8f s = (v8f){0.f,0.f,0.f,0.f,0.f,0.f,0.f,0.f};
#pragma unroll
    for (int kk = 0; kk < HD; kk += 32) {
      int kb = kk + lhalf * 8;
      v16bf a = cat16(*(const v8bf*)&Qs[qrow + lrow][kb],
                      *(const v8bf*)&Qs[qrow + lrow][kb + 16]);
      const bf16_t* kp = Kb + (size_t)(n0 + lrow) * HD + kk + lhalf * 8;
      v16bf b = cat16(*(const v8bf*)kp, *(const v8bf*)(kp + 16));
      s = __builtin_amdgcn_wmma_f32_16x16x32_bf16(false, a, false, b,
                                                  (short)0, s, false, false);
    }
#pragma unroll
    for (int v = 0; v < 8; ++v) {  // lane-local online update (no shuffles)
      float x  = s[v];
      float nm = fmaxf(mrow[v], x);
      srow[v]  = srow[v] * __expf(mrow[v] - nm) + __expf(x - nm);
      mrow[v]  = nm;
    }
  }
  // one associative cross-lane combine over the 16 lanes of each half
#pragma unroll
  for (int v = 0; v < 8; ++v) {
#pragma unroll
    for (int d = 1; d < 16; d <<= 1) {
      float om = __shfl_xor(mrow[v], d, 32);
      float os = __shfl_xor(srow[v], d, 32);
      float nm = fmaxf(mrow[v], om);
      srow[v]  = srow[v] * __expf(mrow[v] - nm) + os * __expf(om - nm);
      mrow[v]  = nm;
    }
  }

  // ---- pass 2: recompute S, P = exp(S-m), out += P*V (no barriers) --------
  v8f oacc[8];
#pragma unroll
  for (int j = 0; j < 8; ++j) oacc[j] = (v8f){0.f,0.f,0.f,0.f,0.f,0.f,0.f,0.f};

  for (int n0 = 0; n0 < SEQ; n0 += 32) {
#pragma unroll
    for (int half = 0; half < 2; ++half) {
      v8f s = (v8f){0.f,0.f,0.f,0.f,0.f,0.f,0.f,0.f};
#pragma unroll
      for (int kk = 0; kk < HD; kk += 32) {
        int kb = kk + lhalf * 8;
        v16bf a = cat16(*(const v8bf*)&Qs[qrow + lrow][kb],
                        *(const v8bf*)&Qs[qrow + lrow][kb + 16]);
        const bf16_t* kp = Kb + (size_t)(n0 + half * 16 + lrow) * HD + kk + lhalf * 8;
        v16bf b = cat16(*(const v8bf*)kp, *(const v8bf*)(kp + 16));
        s = __builtin_amdgcn_wmma_f32_16x16x32_bf16(false, a, false, b,
                                                    (short)0, s, false, false);
      }
#pragma unroll
      for (int v = 0; v < 8; ++v)  // C layout -> Ps[row][key], wave-private
        Ps[wave][lhalf * 8 + v][half * 16 + lrow] = (bf16_t)__expf(s[v] - mrow[v]);
    }
    // same-wave LDS RAW: in-order DS + compiler s_wait_dscnt
    int kb = lhalf * 8;
    v16bf pa = cat16(*(const v8bf*)&Ps[wave][lrow][kb],
                     *(const v8bf*)&Ps[wave][lrow][kb + 16]);
#pragma unroll
    for (int j = 0; j < 8; ++j) {
      const bf16_t* vp = Vtb + (size_t)(j * 16 + lrow) * SEQ + n0 + lhalf * 8;
      v16bf vb = cat16(*(const v8bf*)vp, *(const v8bf*)(vp + 16));
      oacc[j] = __builtin_amdgcn_wmma_f32_16x16x32_bf16(false, pa, false, vb,
                                                        (short)0, oacc[j], false, false);
    }
  }

  // finalize: scale rows by 1/sum, store merged-head bf16
  float inv[8];
#pragma unroll
  for (int v = 0; v < 8; ++v) inv[v] = 1.0f / srow[v];
  int b = bh >> 4, h = bh & (HEADS - 1);
#pragma unroll
  for (int j = 0; j < 8; ++j) {
    int hd = j * 16 + lrow;
#pragma unroll
    for (int v = 0; v < 8; ++v) {
      int l = q0 + qrow + lhalf * 8 + v;
      O[(size_t)(b * SEQ + l) * DIM + h * HD + hd] = (bf16_t)(oacc[j][v] * inv[v]);
    }
  }
}

// ---------------------------------------------------------------------------
extern "C" void kernel_launch(void* const* d_in, const int* in_sizes, int n_in,
                              void* d_out, int out_size, void* d_ws, size_t ws_size,
                              hipStream_t stream) {
  (void)in_sizes; (void)n_in; (void)out_size; (void)ws_size;
  const float* x = (const float*)d_in[0];
  const float* Wp[4] = {(const float*)d_in[1], (const float*)d_in[5],
                        (const float*)d_in[9], (const float*)d_in[13]};
  const float* bp[4] = {(const float*)d_in[2], (const float*)d_in[6],
                        (const float*)d_in[10], (const float*)d_in[14]};
  const float* Bp[4] = {(const float*)d_in[3], (const float*)d_in[7],
                        (const float*)d_in[11], (const float*)d_in[15]};
  const float* Ap[4] = {(const float*)d_in[4], (const float*)d_in[8],
                        (const float*)d_in[12], (const float*)d_in[16]};

  char* ws = (char*)d_ws;
  const size_t SZ_X = (size_t)MTOT * DIM * sizeof(bf16_t);  // 32 MiB
  const size_t SZ_W = (size_t)DIM * DIM * sizeof(bf16_t);   // 8 MiB
  bf16_t* Xb  = (bf16_t*)ws;                       ws += SZ_X;
  bf16_t* WT[4];
  for (int p = 0; p < 4; ++p) { WT[p] = (bf16_t*)ws; ws += SZ_W; }
  bf16_t* Qw = (bf16_t*)ws; ws += SZ_X;
  bf16_t* Kw = (bf16_t*)ws; ws += SZ_X;
  bf16_t* Vw = (bf16_t*)ws; ws += SZ_X;   // transposed per head
  bf16_t* AO = (bf16_t*)ws; ws += SZ_X;

  // 1) x -> bf16
  convert_f32_bf16<<<(MTOT * DIM) / 256, 256, 0, stream>>>(x, Xb, MTOT * DIM);
  // 2) effective swizzled weights
  for (int p = 0; p < 4; ++p)
    prep_weight<<<(DIM * DIM) / 256, 256, 0, stream>>>(Wp[p], Ap[p], Bp[p], WT[p]);
  // 3) q/k/v projections (q pre-scaled by 1/sqrt(hd)); V stored transposed
  dim3 ggrid(DIM / 128, MTOT / 128);
  gemm_bias<0><<<ggrid, 256, 0, stream>>>(Xb, WT[0], bp[0], Qw, nullptr, ATT_SCALE);
  gemm_bias<0><<<ggrid, 256, 0, stream>>>(Xb, WT[1], bp[1], Kw, nullptr, 1.0f);
  gemm_bias<2><<<ggrid, 256, 0, stream>>>(Xb, WT[2], bp[2], Vw, nullptr, 1.0f);
  // 4) attention
  dim3 agrid(SEQ / 128, BATCH * HEADS);
  attention<<<agrid, 256, 0, stream>>>(Qw, Kw, Vw, AO);
  // 5) output projection -> f32 d_out
  gemm_bias<1><<<ggrid, 256, 0, stream>>>(AO, WT[3], bp[3], nullptr, (float*)d_out, 1.0f);
}
